// MyLSTMModel_24739011625486
// MI455X (gfx1250) — compile-verified, matrix operand order checked by reading, not run on hardware
//
#include <hip/hip_runtime.h>
#include <hip/hip_bf16.h>
#include <math.h>

typedef __attribute__((ext_vector_type(16))) _Float16 v16h;
typedef __attribute__((ext_vector_type(8)))  float    v8f;

// ---------------- problem constants ----------------
#define BB 2
#define TT 10
#define HH 32
#define WW 32
#define HWPIX 1024         // 32*32
#define MDIM 2048          // B*H*W
#define MT 128             // MDIM/16

// Layer 1: C=3, F=256, 5x5
#define C1 3
#define F1 256
#define CT1 259
#define KD1 6475           // 25*259
#define KT1 203            // ceil(6475/32)
#define KDP1 6496
#define N1 1024
#define NT1 64
// Layer 2: C=256, F=128, 3x3
#define C2 256
#define F2 128
#define CT2 384
#define KD2 3456           // 9*384 (multiple of 32)
#define KT2 108
#define KDP2 3456
#define N2 512
#define NT2 32
// Layer 3: C=128, F=3, 3x3 (N padded 12 -> 16)
#define C3 128
#define F3 3
#define CT3 131
#define KD3 1179           // 9*131
#define KT3 37
#define KDP3 1184
#define N3REAL 12
#define N3 16
#define NT3 1

static __device__ __forceinline__ float hsig(float x) {
    return fminf(fmaxf(0.2f * x + 0.5f, 0.f), 1.f);
}
static __device__ __forceinline__ float sigm(float x) {
    return 1.f / (1.f + __expf(-x));
}

// Inverse fragment map: 32-wide K index -> (half, elem) of the ISA 7.12.2
// 16-bit fragment layout (identical structure for A rows and B columns).
static __device__ __forceinline__ void kin_to_frag(int kin, int& half, int& e) {
    int lo    = kin & 1;
    int khi   = kin & 30;
    half      = (khi & 8) ? 1 : 0;
    int basep = khi - (half << 3);
    int v     = (basep < 8) ? (basep >> 1) : (4 + ((basep - 16) >> 1));
    e         = (v << 1) | lo;
}

// ---------------- utility kernels ----------------
__global__ void zero_u32(unsigned int* __restrict__ p, int n) {
    int i = blockIdx.x * blockDim.x + threadIdx.x;
    if (i < n) p[i] = 0u;
}

// x: (B,T,H,W,C) f32 -> x16: (T, B*HW, C) f16  (t-major slices for the scan)
__global__ void cvt_x(const float* __restrict__ s, _Float16* __restrict__ d) {
    int i = blockIdx.x * blockDim.x + threadIdx.x;   // over B*T*HW*C = 61440
    if (i >= BB * TT * HWPIX * C1) return;
    int cc = i % C1;
    int r  = i / C1;
    int hw = r % HWPIX; r /= HWPIX;
    int t  = r % TT;
    int b  = r / TT;
    d[(((size_t)t * BB + b) * HWPIX + hw) * C1 + cc] = (_Float16)s[i];
}

// Pack stacked [kernel; recurrent_kernel] (f32, HWIO) into WMMA B-fragment
// tiles: tile (kt,nt) = 32(K)x16(N) stored [lane][elem] contiguously.
__global__ void pack_weights(const float* __restrict__ kw,
                             const float* __restrict__ rkw,
                             _Float16* __restrict__ wp,
                             int Cin, int Fh, int Nreal, int Npad, int NT,
                             int Kdim, int KdimPad) {
    int i = blockIdx.x * blockDim.x + threadIdx.x;
    if (i >= KdimPad * Npad) return;
    int kg = i / Npad;
    int n  = i % Npad;
    int Ctot = Cin + Fh;
    float val = 0.f;
    if (kg < Kdim && n < Nreal) {
        int kk = kg / Ctot;
        int c  = kg % Ctot;
        val = (c < Cin) ? kw[(kk * Cin + c) * Nreal + n]
                        : rkw[(kk * Fh + (c - Cin)) * Nreal + n];
    }
    int kt = kg >> 5, kin = kg & 31;
    int nt = n >> 4,  nin = n & 15;
    int half, e;
    kin_to_frag(kin, half, e);
    wp[(((size_t)kt * NT + nt) << 9) + (((half << 4) + nin) << 4) + e] = (_Float16)val;
}

// ---------------- im2col into A-fragment layout ----------------
template <int KW, int PAD, int CA, int CB, int KDIM, int KT>
__global__ void im2col_pack(const _Float16* __restrict__ xin,   // (M, CA)
                            const _Float16* __restrict__ hin,   // (M, CB)
                            _Float16* __restrict__ apack) {     // MT*KT*512
    constexpr int CTOT = CA + CB;
    constexpr int KDP  = KT * 32;
    int i = blockIdx.x * blockDim.x + threadIdx.x;   // over MDIM*KDP
    if (i >= MDIM * KDP) return;
    int m  = i / KDP;
    int kg = i % KDP;

    _Float16 val = (_Float16)0.f;
    if ((KDIM % 32 == 0) || (kg < KDIM)) {
        int kk  = kg / CTOT;
        int c   = kg % CTOT;
        int kh  = kk / KW;
        int kwp = kk % KW;
        int b   = m >> 10;
        int hw  = m & 1023;
        int sh  = (hw >> 5) + kh - PAD;
        int sw  = (hw & 31) + kwp - PAD;
        if (sh >= 0 && sh < HH && sw >= 0 && sw < WW) {
            int pix = ((b << 5) + sh) * WW + sw;
            val = (c < CA) ? xin[(size_t)pix * CA + c]
                           : hin[(size_t)pix * CB + (c - CA)];
        }
    }
    int mt = m >> 4, ml = m & 15;
    int kt = kg >> 5, kin = kg & 31;
    int half, e;
    kin_to_frag(kin, half, e);
    apack[(((size_t)mt * KT + kt) << 9) + (((half << 4) + ml) << 4) + e] = val;
}

// ---------------- fragment GEMM: double-buffered WMMA pipeline ----------
// One wave computes a (16*MB) x (16*NB) macro-tile. K-tile kt+1 fragments
// are prefetched into a ping-pong register set while kt is consumed, so
// VMEM overlaps the XDL pipe. Buffers carry one tile of slack so the final
// prefetch may overshoot harmlessly.
template <int KT, int NT, int NB, int MB, int N>
__global__ void __launch_bounds__(256)
frag_gemm(const _Float16* __restrict__ apack,
          const _Float16* __restrict__ wp,
          float* __restrict__ g) {
    constexpr int NG  = NT / NB;
    constexpr int MG  = MT / MB;
    constexpr int KTE = KT & ~1;     // even part of trip count
    const int lane = threadIdx.x & 31;
    const int wave = threadIdx.x >> 5;
    const int tile = blockIdx.x * 8 + wave;
    if (tile >= MG * NG) return;                 // wave-uniform
    const int tm0 = (tile / NG) * MB;
    const int tn0 = (tile % NG) * NB;

    const v16h* ap = (const v16h*)apack + (size_t)tm0 * KT * 32 + lane;
    const v16h* bp = (const v16h*)wp + (size_t)tn0 * 32 + lane;

    v8f acc[MB][NB];
#pragma unroll
    for (int i = 0; i < MB; ++i)
#pragma unroll
        for (int j = 0; j < NB; ++j)
            acc[i][j] = (v8f){0.f,0.f,0.f,0.f,0.f,0.f,0.f,0.f};

    v16h aReg[2][MB];
    v16h bReg[2][NB];
#pragma unroll
    for (int i = 0; i < MB; ++i) aReg[0][i] = ap[(size_t)i * KT * 32];
#pragma unroll
    for (int j = 0; j < NB; ++j) bReg[0][j] = bp[(size_t)j * 32];

#pragma unroll 2
    for (int kt = 0; kt < KTE; ++kt) {
        const int cur = kt & 1;
        const int nxt = cur ^ 1;
        // prefetch kt+1 (unconditional; buffers padded with slack)
#pragma unroll
        for (int i = 0; i < MB; ++i)
            aReg[nxt][i] = ap[(size_t)(kt + 1) * 32 + (size_t)i * KT * 32];
#pragma unroll
        for (int j = 0; j < NB; ++j)
            bReg[nxt][j] = bp[(size_t)(kt + 1) * NT * 32 + (size_t)j * 32];
        // consume kt
#pragma unroll
        for (int i = 0; i < MB; ++i)
#pragma unroll
            for (int j = 0; j < NB; ++j)
                acc[i][j] = __builtin_amdgcn_wmma_f32_16x16x32_f16(
                    false, aReg[cur][i], false, bReg[cur][j],
                    (short)0, acc[i][j], false, false);
    }
    if (KT & 1) {   // compile-time epilogue; last tile sits in buffer 0
#pragma unroll
        for (int i = 0; i < MB; ++i)
#pragma unroll
            for (int j = 0; j < NB; ++j)
                acc[i][j] = __builtin_amdgcn_wmma_f32_16x16x32_f16(
                    false, aReg[0][i], false, bReg[0][j],
                    (short)0, acc[i][j], false, false);
    }

    const int hf = lane >> 4;
    const int ml = lane & 15;
#pragma unroll
    for (int i = 0; i < MB; ++i) {
#pragma unroll
        for (int j = 0; j < NB; ++j) {
            const int ncol = (tn0 + j) * 16 + ml;
#pragma unroll
            for (int r = 0; r < 8; ++r) {
                const int mrow = (hf << 3) + r;
                g[(size_t)((tm0 + i) * 16 + mrow) * N + ncol] = acc[i][j][r];
            }
        }
    }
}

// ---------------- fused LSTM cell update (+BN for next layer, +output) ----
__global__ void lstm_update(const float* __restrict__ g, int Ng, int Fch,
                            const float* __restrict__ bias,
                            float* __restrict__ c,
                            _Float16* __restrict__ hprev,
                            _Float16* __restrict__ hbn,  // nullable
                            const float* __restrict__ gam,
                            const float* __restrict__ bet,
                            const float* __restrict__ mu,
                            const float* __restrict__ var,
                            float* __restrict__ out,     // nullable
                            int t, int actSigmoid) {
    int idx = blockIdx.x * blockDim.x + threadIdx.x;   // over M*Fch
    if (idx >= MDIM * Fch) return;
    int pix = idx / Fch;
    int f   = idx % Fch;
    const float* gr = g + (size_t)pix * Ng;
    float gi = gr[f]            + bias[f];
    float gf = gr[Fch + f]      + bias[Fch + f];
    float gc = gr[2 * Fch + f]  + bias[2 * Fch + f];
    float go = gr[3 * Fch + f]  + bias[3 * Fch + f];
    float ii = hsig(gi);
    float ff = hsig(gf);
    float oo = hsig(go);
    float cand = actSigmoid ? sigm(gc) : fmaxf(gc, 0.f);
    float cn = ff * c[idx] + ii * cand;
    c[idx] = cn;
    float hn = oo * (actSigmoid ? sigm(cn) : fmaxf(cn, 0.f));
    hprev[idx] = (_Float16)hn;
    if (hbn) {
        float bn = gam[f] * (hn - mu[f]) * rsqrtf(var[f] + 1e-3f) + bet[f];
        hbn[idx] = (_Float16)bn;
    }
    if (out && t < 5) {
        int b  = pix >> 10;
        int hw = pix & 1023;
        out[(((size_t)b * 5 + t) * HWPIX + hw) * 3 + f] = hn;
    }
}

// ---------------- host orchestration ----------------
extern "C" void kernel_launch(void* const* d_in, const int* in_sizes, int n_in,
                              void* d_out, int out_size, void* d_ws, size_t ws_size,
                              hipStream_t stream) {
    (void)in_sizes; (void)n_in; (void)out_size; (void)ws_size;
    const float* x   = (const float*)d_in[0];
    const float* k1  = (const float*)d_in[1];
    const float* rk1 = (const float*)d_in[2];
    const float* b1  = (const float*)d_in[3];
    const float* g1  = (const float*)d_in[4];
    const float* be1 = (const float*)d_in[5];
    const float* m1  = (const float*)d_in[6];
    const float* v1  = (const float*)d_in[7];
    const float* k2  = (const float*)d_in[8];
    const float* rk2 = (const float*)d_in[9];
    const float* b2  = (const float*)d_in[10];
    const float* g2  = (const float*)d_in[11];
    const float* be2 = (const float*)d_in[12];
    const float* m2  = (const float*)d_in[13];
    const float* v2  = (const float*)d_in[14];
    const float* k3  = (const float*)d_in[15];
    const float* rk3 = (const float*)d_in[16];
    const float* b3  = (const float*)d_in[17];
    float* out = (float*)d_out;

    char* ws = (char*)d_ws;
    size_t off = 0;
    auto alloc = [&](size_t bytes) -> char* {
        char* q = ws + off;
        off = (off + bytes + 255) & ~(size_t)255;
        return q;
    };

    // zeroed-state region first (contiguous -> single zero kernel)
    float*    c1s  = (float*)alloc((size_t)MDIM * F1 * 4);
    float*    c2s  = (float*)alloc((size_t)MDIM * F2 * 4);
    float*    c3s  = (float*)alloc((size_t)MDIM * F3 * 4);
    _Float16* h1p  = (_Float16*)alloc((size_t)MDIM * F1 * 2);
    _Float16* h2p  = (_Float16*)alloc((size_t)MDIM * F2 * 2);
    _Float16* h3p  = (_Float16*)alloc((size_t)MDIM * F3 * 2);
    size_t zeroWords = off / 4;

    // weight / activation packs carry one extra K-tile row of slack for the
    // pipeline's overshooting prefetch
    _Float16* wp1  = (_Float16*)alloc(((size_t)KDP1 * N1 + (size_t)NT1 * 512) * 2);
    _Float16* wp2  = (_Float16*)alloc(((size_t)KDP2 * N2 + (size_t)NT2 * 512) * 2);
    _Float16* wp3  = (_Float16*)alloc(((size_t)KDP3 * N3 + (size_t)NT3 * 512) * 2);
    _Float16* x16  = (_Float16*)alloc((size_t)TT * MDIM * C1 * 2);
    _Float16* h1bn = (_Float16*)alloc((size_t)TT * MDIM * F1 * 2);
    _Float16* h2bn = (_Float16*)alloc((size_t)TT * MDIM * F2 * 2);
    float*    gbuf = (float*)alloc((size_t)MDIM * N1 * 4);
    _Float16* apk  = (_Float16*)alloc(((size_t)MDIM * KDP1 + 1024) * 2);

    const int TB = 256;

    zero_u32<<<(int)((zeroWords + TB - 1) / TB), TB, 0, stream>>>(
        (unsigned int*)ws, (int)zeroWords);
    cvt_x<<<(BB * TT * HWPIX * C1 + TB - 1) / TB, TB, 0, stream>>>(x, x16);
    pack_weights<<<(KDP1 * N1 + TB - 1) / TB, TB, 0, stream>>>(
        k1, rk1, wp1, C1, F1, N1, N1, NT1, KD1, KDP1);
    pack_weights<<<(KDP2 * N2 + TB - 1) / TB, TB, 0, stream>>>(
        k2, rk2, wp2, C2, F2, N2, N2, NT2, KD2, KDP2);
    pack_weights<<<(KDP3 * N3 + TB - 1) / TB, TB, 0, stream>>>(
        k3, rk3, wp3, C3, F3, N3REAL, N3, NT3, KD3, KDP3);

    // ---- layer 1 scan ----
    for (int t = 0; t < TT; ++t) {
        im2col_pack<5, 2, C1, F1, KD1, KT1>
            <<<(MDIM * KDP1 + TB - 1) / TB, TB, 0, stream>>>(
                x16 + (size_t)t * MDIM * C1, h1p, apk);
        frag_gemm<KT1, NT1, 4, 2, N1>
            <<<((MT / 2) * (NT1 / 4)) / 8, 256, 0, stream>>>(apk, wp1, gbuf);
        lstm_update<<<(MDIM * F1) / TB, TB, 0, stream>>>(
            gbuf, N1, F1, b1, c1s, h1p, h1bn + (size_t)t * MDIM * F1,
            g1, be1, m1, v1, nullptr, t, 0);
    }
    // ---- layer 2 scan ----
    for (int t = 0; t < TT; ++t) {
        im2col_pack<3, 1, C2, F2, KD2, KT2>
            <<<(MDIM * KDP2 + TB - 1) / TB, TB, 0, stream>>>(
                h1bn + (size_t)t * MDIM * F1, h2p, apk);
        frag_gemm<KT2, NT2, 4, 2, N2>
            <<<((MT / 2) * (NT2 / 4)) / 8, 256, 0, stream>>>(apk, wp2, gbuf);
        lstm_update<<<(MDIM * F2) / TB, TB, 0, stream>>>(
            gbuf, N2, F2, b2, c2s, h2p, h2bn + (size_t)t * MDIM * F2,
            g2, be2, m2, v2, nullptr, t, 0);
    }
    // ---- layer 3 scan (sigmoid activation, writes first 5 steps) ----
    for (int t = 0; t < TT; ++t) {
        im2col_pack<3, 1, C3, F3, KD3, KT3>
            <<<(MDIM * KDP3 + TB - 1) / TB, TB, 0, stream>>>(
                h2bn + (size_t)t * MDIM * F2, h3p, apk);
        frag_gemm<KT3, NT3, 1, 2, N3>
            <<<((MT / 2) * NT3) / 8, 256, 0, stream>>>(apk, wp3, gbuf);
        lstm_update<<<(MDIM * F3 + TB - 1) / TB, TB, 0, stream>>>(
            gbuf, N3, F3, b3, c3s, h3p, nullptr,
            nullptr, nullptr, nullptr, nullptr, out, t, 1);
    }
}
